// ResidualLSTMBlock_8546984919122
// MI455X (gfx1250) — matrix operand order repaired and checked by reading, not compiled, not run on hardware
//
#include <hip/hip_runtime.h>
#include <hip/hip_bf16.h>

// ---------------------------------------------------------------------------
// Residual LSTM block for MI455X (gfx1250, wave32, WMMA).
// B=32, T=512, H=1024.  out = tanh(lstm_scan(x@W + b, U) + x)
// ---------------------------------------------------------------------------

typedef __attribute__((ext_vector_type(16))) __bf16 v16bf;
typedef __attribute__((ext_vector_type(8)))  __bf16 v8bf;
typedef __attribute__((ext_vector_type(8)))  float  v8f;
typedef __attribute__((ext_vector_type(4)))  float  v4f;

#define B_  32
#define T_  512
#define H_  1024
#define G4  4096   // 4*H

// ---- workspace layout (bytes) ----
#define OFF_XZ   ((size_t)0)                            // xz  f32  [B*T, 4H]   256 MB
#define OFF_XB   (OFF_XZ + (size_t)B_*T_*G4*4)          // x   bf16 [B*T, H]     32 MB
#define OFF_WT   (OFF_XB + (size_t)B_*T_*H_*2)          // W^T bf16 [4H, H]       8 MB
#define OFF_UT   (OFF_WT + (size_t)G4*H_*2)             // U^T bf16 [4H, H]       8 MB
#define OFF_H0   (OFF_UT + (size_t)G4*H_*2)             // h ping f32 [B, H]
#define OFF_H1   (OFF_H0 + (size_t)B_*H_*4)             // h pong f32 [B, H]
#define OFF_C    (OFF_H1 + (size_t)B_*H_*4)             // c      f32 [B, H]

__device__ __forceinline__ float sigmf(float x) {
    return 1.0f / (1.0f + __expf(-x));
}

// A-matrix 16x32 bf16 fragment (CDNA5 ISA, 16-bit A table):
// lane<16: M=lane, K in {0..7, 16..23}; lane>=16: M=lane-16, K in {8..15, 24..31}.
// `p` is already offset by the lane's row and (half*8); fetch +0 and +16.
__device__ __forceinline__ v16bf load_fragA_bf16(const __bf16* p) {
    v8bf lo = *(const v8bf*)(p);        // global_load_b128
    v8bf hi = *(const v8bf*)(p + 16);   // global_load_b128
    v16bf f;
#pragma unroll
    for (int i = 0; i < 8; ++i) { f[i] = lo[i]; f[8 + i] = hi[i]; }
    return f;
}

// Same A fragment but source rows are fp32 (the live h state): convert on the fly.
__device__ __forceinline__ v16bf load_fragA_f32(const float* p) {
    v4f a0 = *(const v4f*)(p + 0);
    v4f a1 = *(const v4f*)(p + 4);
    v4f b0 = *(const v4f*)(p + 16);
    v4f b1 = *(const v4f*)(p + 20);
    v16bf f;
#pragma unroll
    for (int i = 0; i < 4; ++i) {
        f[i]      = (__bf16)a0[i];
        f[4 + i]  = (__bf16)a1[i];
        f[8 + i]  = (__bf16)b0[i];
        f[12 + i] = (__bf16)b1[i];
    }
    return f;
}

// B-matrix 32x16 bf16 fragment (same K order as A: lane<16 -> N=lane,
// K in {0..7, 16..23}; lane>=16 -> K in {8..15, 24..31}). `p` already offset by the lane's
// column-row in the transposed weight and (half*8); fetch +0 and +16.
__device__ __forceinline__ v16bf load_fragB_bf16(const __bf16* p) {
    v8bf lo = *(const v8bf*)(p);        // K base+0..7
    v8bf hi = *(const v8bf*)(p + 16);   // K base+16..23
    v16bf f;
#pragma unroll
    for (int i = 0; i < 8; ++i) { f[i] = lo[i]; f[8 + i] = hi[i]; }
    return f;
}

#define WMMA_BF16(A, Bm, C) \
    __builtin_amdgcn_wmma_f32_16x16x32_bf16(false, (A), false, (Bm), (short)0, (C), false, false)

// ---------------------------------------------------------------------------
// 1) Transpose+convert W and U -> bf16 [4H, H].
// ---------------------------------------------------------------------------
__global__ void convert_wu(const float* __restrict__ W, const float* __restrict__ U,
                           __bf16* __restrict__ Wt, __bf16* __restrict__ Ut) {
    int idx = blockIdx.x * blockDim.x + threadIdx.x;   // over H*4H elements
    int k = idx / G4;            // 0..H-1
    int n = idx % G4;            // 0..4H-1
    Wt[(size_t)n * H_ + k] = (__bf16)W[idx];
    Ut[(size_t)n * H_ + k] = (__bf16)U[idx];
}

// ---------------------------------------------------------------------------
// 2) x -> bf16 (rows already K-contiguous for A fragments)
// ---------------------------------------------------------------------------
__global__ void convert_x(const float* __restrict__ x, __bf16* __restrict__ xb) {
    int idx = blockIdx.x * blockDim.x + threadIdx.x;   // over B*T*H
    xb[idx] = (__bf16)x[idx];
}

// ---------------------------------------------------------------------------
// 3) zero h0 and c
// ---------------------------------------------------------------------------
__global__ void init_state(float* __restrict__ h0, float* __restrict__ c) {
    int idx = blockIdx.x * blockDim.x + threadIdx.x;   // over B*H
    h0[idx] = 0.0f;
    c[idx]  = 0.0f;
}

// ---------------------------------------------------------------------------
// 4) xz = x @ W + b     (M=16384, N=4096, K=1024)
//    one wave -> 16x64 tile; depth-1 staggered pipeline (each WMMA consumes a
//    load issued one slot earlier; only A and the first B of the next K-step
//    are pre-rotated -> no spills).  Non-temporal xz stores keep L2 for U.
// ---------------------------------------------------------------------------
__global__ void gemm_xz(const __bf16* __restrict__ xb, const __bf16* __restrict__ Wt,
                        const float* __restrict__ bias, float* __restrict__ xz) {
    const int lane = threadIdx.x & 31;
    const int wid  = (blockIdx.x * blockDim.x + threadIdx.x) >> 5; // 0..65535
    const int m0   = (wid >> 6) * 16;   // M tile
    const int n0   = (wid & 63) * 64;   // group of 4 N tiles

    const int ln   = lane & 15;
    const int half = lane >> 4;

    v8f acc0 = {}, acc1 = {}, acc2 = {}, acc3 = {};

    const __bf16* aRow  = xb + (size_t)(m0 + ln) * H_ + half * 8;            // A layout
    const __bf16* bRow0 = Wt + (size_t)(n0 + 0 * 16 + ln) * H_ + half * 8;  // B layout
    const __bf16* bRow1 = Wt + (size_t)(n0 + 1 * 16 + ln) * H_ + half * 8;
    const __bf16* bRow2 = Wt + (size_t)(n0 + 2 * 16 + ln) * H_ + half * 8;
    const __bf16* bRow3 = Wt + (size_t)(n0 + 3 * 16 + ln) * H_ + half * 8;

    v16bf a  = load_fragA_bf16(aRow);
    v16bf bn = load_fragB_bf16(bRow0);

    int k0 = 0;
    for (; k0 < H_ - 32; k0 += 32) {
        v16bf b0c = bn;
        v16bf b1c = load_fragB_bf16(bRow1 + k0);
        acc0 = WMMA_BF16(a, b0c, acc0);
        v16bf b2c = load_fragB_bf16(bRow2 + k0);
        acc1 = WMMA_BF16(a, b1c, acc1);
        v16bf b3c = load_fragB_bf16(bRow3 + k0);
        acc2 = WMMA_BF16(a, b2c, acc2);
        v16bf an = load_fragA_bf16(aRow + k0 + 32);
        bn = load_fragB_bf16(bRow0 + k0 + 32);
        __builtin_prefetch(bRow0 + k0 + 96, 0, 1);      // global_prefetch_b8
        acc3 = WMMA_BF16(a, b3c, acc3);
        a = an;
    }
    {   // peeled tail: k0 == H_-32, no next-iteration loads
        v16bf b1c = load_fragB_bf16(bRow1 + k0);
        v16bf b2c = load_fragB_bf16(bRow2 + k0);
        v16bf b3c = load_fragB_bf16(bRow3 + k0);
        acc0 = WMMA_BF16(a, bn,  acc0);
        acc1 = WMMA_BF16(a, b1c, acc1);
        acc2 = WMMA_BF16(a, b2c, acc2);
        acc3 = WMMA_BF16(a, b3c, acc3);
    }

    // C/D layout: VGPR r, lanes 0-15 -> M=r ; lanes 16-31 -> M=r+8 ; N = lane&15
#pragma unroll
    for (int s = 0; s < 4; ++s) {
        const v8f acc = (s == 0) ? acc0 : (s == 1) ? acc1 : (s == 2) ? acc2 : acc3;
        const int n  = n0 + s * 16 + ln;
        const float bv = bias[n];
#pragma unroll
        for (int r = 0; r < 8; ++r) {
            const int m = m0 + r + (half << 3);
            __builtin_nontemporal_store(acc[r] + bv, &xz[(size_t)m * G4 + n]);
        }
    }
}

// ---------------------------------------------------------------------------
// 5) One LSTM timestep:  z = xz[:,t,:] + h_prev @ U ; gates ; c,h ;
//    out[:,t,:] = tanh(h + x[:,t,:]).
//
//    128 blocks x 128 threads. Block = one (batch-half m0, 16 hidden cols j0)
//    tile of all four gates; its 4 waves each take a K-chunk of 256 (512
//    waves/step, U traffic unchanged and L2-resident in bf16). Partial f32
//    accumulators are reduced through LDS; fused gate/c/h/residual epilogue.
//    h ping-pongs across the 512 stream-ordered launches.
// ---------------------------------------------------------------------------
__global__ void lstm_step(const float* __restrict__ xz, const __bf16* __restrict__ Ut,
                          const float* __restrict__ h_prev, float* __restrict__ h_next,
                          float* __restrict__ c, const float* __restrict__ x,
                          float* __restrict__ out, int t) {
    __shared__ float red[4][4][256];     // [gate][k-chunk wave][tile element]

    const int lane = threadIdx.x & 31;
    const int wk   = threadIdx.x >> 5;   // 0..3 : K chunk of 256
    const int m0   = (blockIdx.x & 1) * 16;    // batch-half
    const int j0   = (blockIdx.x >> 1) * 16;   // hidden column tile

    const int ln   = lane & 15;
    const int half = lane >> 4;

    v8f accI = {}, accF = {}, accG = {}, accO = {};

    const int kbase = wk * 256;
    const float*  aRow = h_prev + (size_t)(m0 + ln) * H_ + kbase + half * 8;
    const __bf16* uI = Ut + (size_t)(0 * H_ + j0 + ln) * H_ + kbase + half * 8;
    const __bf16* uF = Ut + (size_t)(1 * H_ + j0 + ln) * H_ + kbase + half * 8;
    const __bf16* uG = Ut + (size_t)(2 * H_ + j0 + ln) * H_ + kbase + half * 8;
    const __bf16* uO = Ut + (size_t)(3 * H_ + j0 + ln) * H_ + kbase + half * 8;

    v16bf a  = load_fragA_f32(aRow);     // fp32 h -> bf16 fragment
    v16bf bn = load_fragB_bf16(uI);

    int k0 = 0;
    for (; k0 < 256 - 32; k0 += 32) {
        v16bf bic = bn;
        v16bf bfc = load_fragB_bf16(uF + k0);
        accI = WMMA_BF16(a, bic, accI);
        v16bf bgc = load_fragB_bf16(uG + k0);
        accF = WMMA_BF16(a, bfc, accF);
        v16bf boc = load_fragB_bf16(uO + k0);
        accG = WMMA_BF16(a, bgc, accG);
        v16bf an = load_fragA_f32(aRow + k0 + 32);
        bn = load_fragB_bf16(uI + k0 + 32);
        __builtin_prefetch(uI + k0 + 64, 0, 1);
        accO = WMMA_BF16(a, boc, accO);
        a = an;
    }
    {   // peeled tail: k0 == 224
        v16bf bfc = load_fragB_bf16(uF + k0);
        v16bf bgc = load_fragB_bf16(uG + k0);
        v16bf boc = load_fragB_bf16(uO + k0);
        accI = WMMA_BF16(a, bn,  accI);
        accF = WMMA_BF16(a, bfc, accF);
        accG = WMMA_BF16(a, bgc, accG);
        accO = WMMA_BF16(a, boc, accO);
    }

    // dump partials: element e = (row-in-tile)*16 + col
#pragma unroll
    for (int r = 0; r < 8; ++r) {
        const int e = ((r + (half << 3)) << 4) | ln;
        red[0][wk][e] = accI[r];
        red[1][wk][e] = accF[r];
        red[2][wk][e] = accG[r];
        red[3][wk][e] = accO[r];
    }
    __syncthreads();

    // fused reduction + gate + state-update + residual epilogue:
    // 128 threads x 2 elements cover the 256-element tile exactly once.
#pragma unroll
    for (int q = 0; q < 2; ++q) {
        const int e = threadIdx.x * 2 + q;
        const int m = m0 + (e >> 4);         // batch index
        const int j = j0 + (e & 15);         // hidden index

        const size_t zrow = ((size_t)m * T_ + t) * G4;
        const float zi = red[0][0][e] + red[0][1][e] + red[0][2][e] + red[0][3][e]
                         + __builtin_nontemporal_load(&xz[zrow + 0 * H_ + j]);
        const float zf = red[1][0][e] + red[1][1][e] + red[1][2][e] + red[1][3][e]
                         + __builtin_nontemporal_load(&xz[zrow + 1 * H_ + j]);
        const float zg = red[2][0][e] + red[2][1][e] + red[2][2][e] + red[2][3][e]
                         + __builtin_nontemporal_load(&xz[zrow + 2 * H_ + j]);
        const float zo = red[3][0][e] + red[3][1][e] + red[3][2][e] + red[3][3][e]
                         + __builtin_nontemporal_load(&xz[zrow + 3 * H_ + j]);

        const float ig = sigmf(zi);
        const float fg = sigmf(zf);
        const float gg = tanhf(zg);
        const float og = sigmf(zo);

        const size_t sj = (size_t)m * H_ + j;
        const float cn = fg * c[sj] + ig * gg;   // element owned by this block only
        const float hn = og * tanhf(cn);
        c[sj]      = cn;
        h_next[sj] = hn;

        const size_t oi = ((size_t)m * T_ + t) * H_ + j;
        out[oi] = tanhf(hn + x[oi]);             // residual add + tanh epilogue
    }
}

// ---------------------------------------------------------------------------
extern "C" void kernel_launch(void* const* d_in, const int* in_sizes, int n_in,
                              void* d_out, int out_size, void* d_ws, size_t ws_size,
                              hipStream_t stream) {
    (void)in_sizes; (void)n_in; (void)out_size; (void)ws_size;

    const float* x    = (const float*)d_in[0];   // [B, T, H]
    const float* W    = (const float*)d_in[1];   // [H, 4H]
    const float* U    = (const float*)d_in[2];   // [H, 4H]
    const float* bias = (const float*)d_in[3];   // [4H]
    float* out = (float*)d_out;                  // [B, T, H]

    char* ws = (char*)d_ws;
    float*  xz = (float*)(ws + OFF_XZ);
    __bf16* xb = (__bf16*)(ws + OFF_XB);
    __bf16* Wt = (__bf16*)(ws + OFF_WT);
    __bf16* Ut = (__bf16*)(ws + OFF_UT);
    float*  h0 = (float*)(ws + OFF_H0);
    float*  h1 = (float*)(ws + OFF_H1);
    float*  c  = (float*)(ws + OFF_C);

    // 1) W,U -> bf16 transposed  (H*4H = 4,194,304 elements)
    convert_wu<<<(H_ * G4) / 256, 256, 0, stream>>>(W, U, Wt, Ut);
    // 2) x -> bf16               (B*T*H = 16,777,216 elements)
    convert_x<<<(B_ * T_ * H_) / 256, 256, 0, stream>>>(x, xb);
    // 3) h0 = c = 0
    init_state<<<(B_ * H_) / 256, 256, 0, stream>>>(h0, c);
    // 4) xz = x @ W + b : 65536 waves, 8 waves/block
    gemm_xz<<<8192, 256, 0, stream>>>(xb, Wt, bias, xz);
    // 5) sequential scan: 512 step launches, h ping-pong, 512 waves/step
    for (int t = 0; t < T_; ++t) {
        float* hp = (t & 1) ? h1 : h0;
        float* hn = (t & 1) ? h0 : h1;
        lstm_step<<<128, 128, 0, stream>>>(xz, Ut, hp, hn, c, x, out, t);
    }
}
